// Model_57638461112382
// MI455X (gfx1250) — matrix operand order repaired and checked by reading, not compiled, not run on hardware
//
#include <hip/hip_runtime.h>
#include <hip/hip_bf16.h>
#include <math.h>

// ---------------------------------------------------------------------------
// Spike-slab Bayesian linear layer, MI455X (gfx1250) version.
//   m = x @ (spike*mu).T ; v = x^2 @ var_pre.T ; out = [m+eps*sqrt(v), m-...]
// Two bf16 WMMA GEMMs fused over one pass of x (512 MB read once),
// weights pre-converted to bf16 (L2-resident), f32 accumulation.
// ---------------------------------------------------------------------------

typedef __attribute__((ext_vector_type(16))) __bf16 v16bf;
typedef __attribute__((ext_vector_type(8)))  __bf16 v8bf;
typedef __attribute__((ext_vector_type(4)))  __bf16 v4bf;
typedef __attribute__((ext_vector_type(8)))  float  v8f;
typedef __attribute__((ext_vector_type(4)))  float  v4f;

#define B_SZ   4096
#define D_IN   32768
#define D_OUT  128
#define OUT_SZ (B_SZ * D_OUT)          // 524288 per output slab
#define ALPHA_C 0.1f
#define EPS_C   1e-12f
#define KCH   128                      // K staged per LDS round
#define LDSW  136                      // padded LDS row stride (bf16) -> conflict-free

// ---------------------------------------------------------------------------
// Kernel 1: weights -> bf16 mean_pre / var_pre, plus KL partial sums.
// 4096 blocks x 256 threads x 4 elements = 4M elements.
// ---------------------------------------------------------------------------
__global__ __launch_bounds__(256) void prep_kernel(
    const float* __restrict__ mu, const float* __restrict__ sg,
    const float* __restrict__ sp, const float* __restrict__ pr,
    __bf16* __restrict__ meanw, __bf16* __restrict__ varw,
    float* __restrict__ klp)
{
    const int tid = threadIdx.x;
    const size_t base = ((size_t)blockIdx.x * 256 + tid) * 4;
    v4f m4 = *(const v4f*)(mu + base);
    v4f s4 = *(const v4f*)(sg + base);
    v4f p4 = *(const v4f*)(sp + base);
    v4f r4 = *(const v4f*)(pr + base);
    v4bf mo, vo;
    float kl = 0.f;
#pragma unroll
    for (int j = 0; j < 4; ++j) {
        float muv = m4[j];
        float sig = fmaxf(s4[j], EPS_C);                        // relu + clamp(eps)
        float spk = fminf(fmaxf(p4[j], 1e-6f), 1.f - 1e-6f);
        float pv  = r4[j] * r4[j];
        float mean = spk * muv;
        float var  = spk * (sig * sig + muv * muv) - mean * mean; // >= 0
        mo[j] = (__bf16)mean;
        vo[j] = (__bf16)var;
        float inv_pv = 1.f / pv;
        float g = -0.5f * spk * (1.f + 2.f * logf(sig) - logf(pv)
                                 - muv * muv * inv_pv - sig * sig * inv_pv);
        float skl = (1.f - spk) * logf((1.f - spk) / (1.f - ALPHA_C))
                  + spk * logf(spk / ALPHA_C);
        kl += g + skl;
    }
    *(v4bf*)(meanw + base) = mo;
    *(v4bf*)(varw + base)  = vo;

    __shared__ float red[256];
    red[tid] = kl;
    __syncthreads();
    for (int s = 128; s > 0; s >>= 1) {
        if (tid < s) red[tid] += red[tid + s];
        __syncthreads();
    }
    if (tid == 0) klp[blockIdx.x] = red[0];
}

// ---------------------------------------------------------------------------
// Kernel 2: fused dual GEMM (m and v) with v_wmma_f32_16x16x32_bf16.
// Workgroup: 8 waves; wave w owns M rows [m0, m0+16) x all 128 N columns.
// K split across blockIdx.y into f32 partials (deterministic reduction later).
// Weight tiles staged in LDS so 8 waves share one L2 fetch.
// ---------------------------------------------------------------------------
__global__ __launch_bounds__(256) void gemm_kernel(
    const float* __restrict__ x,
    const __bf16* __restrict__ meanw, const __bf16* __restrict__ varw,
    float* __restrict__ partM, float* __restrict__ partV, int klen)
{
    __shared__ __bf16 smM[128 * LDSW];
    __shared__ __bf16 smV[128 * LDSW];

    const int tid   = threadIdx.x;
    const int wave  = tid >> 5;
    const int lane  = tid & 31;
    const int lhalf = lane >> 4;           // 0: lanes 0-15, 1: lanes 16-31
    const int l15   = lane & 15;
    const int m0    = blockIdx.x * 128 + wave * 16;
    const int ky    = blockIdx.y;
    const int kbeg  = ky * klen;

    v8f zero = {};
    v8f accM[8], accV[8];
#pragma unroll
    for (int nt = 0; nt < 8; ++nt) { accM[nt] = zero; accV[nt] = zero; }

    // cooperative LDS staging: 2 threads per weight row, 64 bf16 (128B) each
    const int ldn = tid >> 1;
    const int ldh = (tid & 1) * 64;

    const float* xrow = x + (size_t)(m0 + l15) * D_IN;

    for (int kc = kbeg; kc < kbeg + klen; kc += KCH) {
#pragma unroll
        for (int j = 0; j < 8; ++j) {
            const size_t goff = (size_t)ldn * D_IN + (size_t)kc + ldh + j * 8;
            *(v8bf*)&smM[ldn * LDSW + ldh + j * 8] = *(const v8bf*)(meanw + goff);
            *(v8bf*)&smV[ldn * LDSW + ldh + j * 8] = *(const v8bf*)(varw  + goff);
        }
        __syncthreads();

#pragma unroll
        for (int s = 0; s < KCH / 32; ++s) {
            const int k0 = kc + s * 32;
            // A layout (16-bit, 16x32): lanes 0-15 hold K{0..7,16..23},
            // lanes 16-31 hold K{8..15,24..31}, M = lane&15.
            const int kA1 = k0 + lhalf * 8;
            v4f x0 = *(const v4f*)(xrow + kA1);
            v4f x1 = *(const v4f*)(xrow + kA1 + 4);
            v4f x2 = *(const v4f*)(xrow + kA1 + 16);
            v4f x3 = *(const v4f*)(xrow + kA1 + 20);
            v16bf a, a2;
#pragma unroll
            for (int j = 0; j < 4; ++j) {
                a[j]      = (__bf16)x0[j];  a2[j]      = (__bf16)(x0[j] * x0[j]);
                a[4 + j]  = (__bf16)x1[j];  a2[4 + j]  = (__bf16)(x1[j] * x1[j]);
                a[8 + j]  = (__bf16)x2[j];  a2[8 + j]  = (__bf16)(x2[j] * x2[j]);
                a[12 + j] = (__bf16)x3[j];  a2[12 + j] = (__bf16)(x3[j] * x3[j]);
            }
            // B layout (32x16): lane = N column, lanes 0-15 K{0..7,16..23},
            // lanes 16-31 K{8..15,24..31} (matches A).
            const int koff = s * 32 + lhalf * 8;
#pragma unroll
            for (int nt = 0; nt < 8; ++nt) {
                const int n = nt * 16 + l15;
                v8bf blo = *(const v8bf*)&smM[n * LDSW + koff];
                v8bf bhi = *(const v8bf*)&smM[n * LDSW + koff + 16];
                v16bf bM = __builtin_shufflevector(blo, bhi,
                    0,1,2,3,4,5,6,7,8,9,10,11,12,13,14,15);
                blo = *(const v8bf*)&smV[n * LDSW + koff];
                bhi = *(const v8bf*)&smV[n * LDSW + koff + 16];
                v16bf bV = __builtin_shufflevector(blo, bhi,
                    0,1,2,3,4,5,6,7,8,9,10,11,12,13,14,15);
                accM[nt] = __builtin_amdgcn_wmma_f32_16x16x32_bf16(
                    false, a,  false, bM, (short)0, accM[nt], false, false);
                accV[nt] = __builtin_amdgcn_wmma_f32_16x16x32_bf16(
                    false, a2, false, bV, (short)0, accV[nt], false, false);
            }
        }
        __syncthreads();
    }

    // C/D layout: VGPR r -> lanes 0-15: (M=r, N=lane), lanes 16-31: (M=8+r).
    const int rbase = lhalf * 8;
    float* pM = partM + (size_t)ky * OUT_SZ;
    float* pV = partV + (size_t)ky * OUT_SZ;
#pragma unroll
    for (int nt = 0; nt < 8; ++nt) {
#pragma unroll
        for (int r = 0; r < 8; ++r) {
            const size_t o = (size_t)(m0 + rbase + r) * D_OUT + nt * 16 + l15;
            pM[o] = accM[nt][r];
            pV[o] = accV[nt][r];
        }
    }
}

// ---------------------------------------------------------------------------
// Kernel 3: deterministic K-partial reduction + epilogue.
// ---------------------------------------------------------------------------
__global__ __launch_bounds__(256) void reduce_kernel(
    const float* __restrict__ partM, const float* __restrict__ partV,
    const float* __restrict__ eps, float* __restrict__ out, int KS)
{
    const int i = blockIdx.x * 256 + threadIdx.x;
    float m = 0.f, v = 0.f;
    for (int ky = 0; ky < KS; ++ky) {
        m += partM[(size_t)ky * OUT_SZ + i];
        v += partV[(size_t)ky * OUT_SZ + i];
    }
    const float s = eps[i] * sqrtf(fmaxf(v, 0.f));
    out[i]          = m + s;
    out[OUT_SZ + i] = m - s;
}

// ---------------------------------------------------------------------------
// Kernel 4: final KL reduction (4096 partials -> scalar).
// ---------------------------------------------------------------------------
__global__ __launch_bounds__(256) void klfin_kernel(
    const float* __restrict__ klp, float* __restrict__ out)
{
    __shared__ float red[256];
    float s = 0.f;
    for (int i = threadIdx.x; i < 4096; i += 256) s += klp[i];
    red[threadIdx.x] = s;
    __syncthreads();
    for (int st = 128; st > 0; st >>= 1) {
        if (threadIdx.x < st) red[threadIdx.x] += red[threadIdx.x + st];
        __syncthreads();
    }
    if (threadIdx.x == 0) out[2 * OUT_SZ] = red[0] / 4096.0f;
}

// ---------------------------------------------------------------------------
// Fallback (only if workspace too small): direct f32 compute, no workspace.
// ---------------------------------------------------------------------------
__global__ __launch_bounds__(256) void fb_out_kernel(
    const float* __restrict__ x, const float* __restrict__ mu,
    const float* __restrict__ sg, const float* __restrict__ sp,
    const float* __restrict__ eps, float* __restrict__ out)
{
    const int i = blockIdx.x * 256 + threadIdx.x;
    const int b = i >> 7, n = i & 127;
    const float* xr  = x  + (size_t)b * D_IN;
    const float* mur = mu + (size_t)n * D_IN;
    const float* sgr = sg + (size_t)n * D_IN;
    const float* spr = sp + (size_t)n * D_IN;
    float m = 0.f, v = 0.f;
    for (int k = 0; k < D_IN; ++k) {
        float xv  = xr[k];
        float sig = fmaxf(sgr[k], EPS_C);
        float spk = fminf(fmaxf(spr[k], 1e-6f), 1.f - 1e-6f);
        float mean = spk * mur[k];
        float var  = spk * (sig * sig + mur[k] * mur[k]) - mean * mean;
        m += xv * mean;
        v += xv * xv * var;
    }
    const float s = eps[i] * sqrtf(fmaxf(v, 0.f));
    out[i] = m + s; out[OUT_SZ + i] = m - s;
}

__global__ __launch_bounds__(256) void fb_kl_kernel(
    const float* __restrict__ mu, const float* __restrict__ sg,
    const float* __restrict__ sp, const float* __restrict__ pr,
    float* __restrict__ out)
{
    float kl = 0.f;
    for (size_t i = threadIdx.x; i < (size_t)D_OUT * D_IN; i += 256) {
        float muv = mu[i];
        float sig = fmaxf(sg[i], EPS_C);
        float spk = fminf(fmaxf(sp[i], 1e-6f), 1.f - 1e-6f);
        float pv  = pr[i] * pr[i];
        kl += -0.5f * spk * (1.f + 2.f * logf(sig) - logf(pv)
                             - muv * muv / pv - sig * sig / pv)
            + (1.f - spk) * logf((1.f - spk) / (1.f - ALPHA_C))
            + spk * logf(spk / ALPHA_C);
    }
    __shared__ float red[256];
    red[threadIdx.x] = kl;
    __syncthreads();
    for (int st = 128; st > 0; st >>= 1) {
        if (threadIdx.x < st) red[threadIdx.x] += red[threadIdx.x + st];
        __syncthreads();
    }
    if (threadIdx.x == 0) out[2 * OUT_SZ] = red[0] / 4096.0f;
}

// ---------------------------------------------------------------------------
extern "C" void kernel_launch(void* const* d_in, const int* in_sizes, int n_in,
                              void* d_out, int out_size, void* d_ws, size_t ws_size,
                              hipStream_t stream)
{
    const float* x   = (const float*)d_in[0];
    const float* wmu = (const float*)d_in[1];
    const float* sg  = (const float*)d_in[2];
    const float* sp  = (const float*)d_in[3];
    const float* pr  = (const float*)d_in[4];
    const float* eps = (const float*)d_in[5];
    float* out = (float*)d_out;

    const size_t W = (size_t)D_OUT * D_IN;           // 4M weight elements
    char* ws = (char*)d_ws;
    const size_t off_mean = 0;                       // 8 MB bf16
    const size_t off_var  = W * 2;                   // 8 MB bf16
    const size_t off_klp  = W * 4;                   // 16 KB f32 partials
    const size_t off_part = off_klp + 4096 * 4;
    const size_t unit = (size_t)OUT_SZ * 4;          // 2 MB per partial slab

    // pick largest K-split that fits the workspace (parallelism vs footprint)
    int KS = 0;
    const int cands[5] = {16, 8, 4, 2, 1};
    for (int c = 0; c < 5; ++c)
        if (off_part + (size_t)cands[c] * unit * 2 <= ws_size) { KS = cands[c]; break; }

    if (KS == 0) {   // workspace too small: slow but correct fallback
        fb_kl_kernel<<<1, 256, 0, stream>>>(wmu, sg, sp, pr, out);
        fb_out_kernel<<<OUT_SZ / 256, 256, 0, stream>>>(x, wmu, sg, sp, eps, out);
        return;
    }

    __bf16* meanw = (__bf16*)(ws + off_mean);
    __bf16* varw  = (__bf16*)(ws + off_var);
    float*  klp   = (float*)(ws + off_klp);
    float*  partM = (float*)(ws + off_part);
    float*  partV = partM + (size_t)KS * OUT_SZ;

    prep_kernel<<<4096, 256, 0, stream>>>(wmu, sg, sp, pr, meanw, varw, klp);
    dim3 grid(B_SZ / 128, KS);
    gemm_kernel<<<grid, 256, 0, stream>>>(x, meanw, varw, partM, partV, D_IN / KS);
    reduce_kernel<<<OUT_SZ / 256, 256, 0, stream>>>(partM, partV, eps, out, KS);
    klfin_kernel<<<1, 256, 0, stream>>>(klp, out);
}
